// DiyTransformer_29489245454473
// MI455X (gfx1250) — compile-verified, matrix operand order checked by reading, not run on hardware
//
#include <hip/hip_runtime.h>

// ---------------------------------------------------------------------------
// MI455X (gfx1250) fused multi-head attention, bf16 WMMA path.
//   x:[2,2048,1024] f32; W*:[1024,1024] f32; b*:[1024] f32; out:[2,2048,1024] f32
// ---------------------------------------------------------------------------

typedef __bf16 bf16;
typedef __attribute__((ext_vector_type(16))) __bf16 v16bf;
typedef __attribute__((ext_vector_type(8)))  float  v8f;
typedef __attribute__((ext_vector_type(4)))  unsigned int v4u;
typedef __attribute__((ext_vector_type(4)))  int v4i;
typedef __attribute__((address_space(1))) v4i g_v4i;   // global
typedef __attribute__((address_space(3))) v4i l_v4i;   // LDS

#if defined(__has_builtin)
#if __has_builtin(__builtin_amdgcn_global_load_async_to_lds_b128)
#define HAVE_ASYNC_LDS 1
#endif
#endif

#define WMMA_BF16(a, b, c) \
    __builtin_amdgcn_wmma_f32_16x16x32_bf16(false, (a), false, (b), (short)0, (c), false, false)

// Load a 16x32 bf16 A/B fragment (8 VGPRs/lane) per ISA 7.12.2:
//   lanes 0-15 : K = k0+0..7  and k0+16..23   (rows/cols = lane)
//   lanes 16-31: K = k0+8..15 and k0+24..31
static __device__ __forceinline__ v16bf load_frag(const bf16* base, int ld, int row0,
                                                  int k0, int lane) {
    const int l16 = lane & 15, hi = lane >> 4;
    const bf16* p = base + (size_t)(row0 + l16) * (size_t)ld + k0 + hi * 8;
    union { v4u u[2]; v16bf v; } un;
    un.u[0] = *(const v4u*)p;        // 16B chunk: K = k0 + hi*8 .. +7
    un.u[1] = *(const v4u*)(p + 16); // 16B chunk: K = k0 + 16 + hi*8 .. +7
    return un.v;
}

static __device__ __forceinline__ float red_max16(float x) {
    #pragma unroll
    for (int m = 1; m < 16; m <<= 1) x = fmaxf(x, __shfl_xor(x, m, 32));
    return x;
}
static __device__ __forceinline__ float red_sum16(float x) {
    #pragma unroll
    for (int m = 1; m < 16; m <<= 1) x += __shfl_xor(x, m, 32);
    return x;
}

// ---------------------------------------------------------------------------
__global__ void mha_cvt_bf16(const float* __restrict__ in, bf16* __restrict__ out, int n) {
    int i = (blockIdx.x * blockDim.x + threadIdx.x) * 8;
    if (i < n) {
        const float4 f0 = *(const float4*)(in + i);
        const float4 f1 = *(const float4*)(in + i + 4);
        union { bf16 h[8]; v4u u; } un;
        un.h[0] = (bf16)f0.x; un.h[1] = (bf16)f0.y; un.h[2] = (bf16)f0.z; un.h[3] = (bf16)f0.w;
        un.h[4] = (bf16)f1.x; un.h[5] = (bf16)f1.y; un.h[6] = (bf16)f1.z; un.h[7] = (bf16)f1.w;
        *(v4u*)(out + i) = un.u;
    }
}

// ---------------------------------------------------------------------------
// Stage a 128(cols) x 32(k) bf16 B-tile into LDS: 8 KB = 512 x 16B chunks,
// 2 chunks per thread, via async global->LDS copies (ASYNCcnt) when available.
static __device__ __forceinline__ void stage_B(const bf16* __restrict__ W, bf16* ldsbuf,
                                               int nblk, int k, int K, int tid) {
    #pragma unroll
    for (int i = 0; i < 2; ++i) {
        const int c = tid + i * 256;
        const int col = c >> 2, kc = c & 3;
        const bf16* src = W + (size_t)(nblk + col) * K + k + kc * 8;
        bf16* dst = ldsbuf + col * 32 + kc * 8;
#if defined(HAVE_ASYNC_LDS)
        __builtin_amdgcn_global_load_async_to_lds_b128((g_v4i*)src, (l_v4i*)dst, 0, 0);
#else
        *(v4u*)dst = *(const v4u*)src;   // fallback: load + ds_store_b128
#endif
    }
}

static __device__ __forceinline__ void wait_async_stage() {
#if defined(HAVE_ASYNC_LDS)
    asm volatile("s_wait_asynccnt 0x0" ::: "memory");
#endif
}

// ---------------------------------------------------------------------------
// Y = A[M,K](bf16) * W[N,K](bf16; B column n == W row n) + bias
// mode 0: bf16 row-major [M,N]   mode 1: bf16 transposed [N,M]   mode 2: f32 [M,N]
// Block 256 thr = 8 waves as 4(M) x 2(N); block tile 128x128; wave tile 32x64.
// B tile double-buffered in LDS via async copies; A direct from global, pipelined.
__global__ void mha_gemm_wmma(const bf16* __restrict__ A, const bf16* __restrict__ W,
                              const float* __restrict__ bias, void* __restrict__ out,
                              int M, int N, int K, int mode) {
    __shared__ alignas(16) bf16 ldsB[2][128 * 32];   // 2 x 8 KB

    const int tid  = threadIdx.x;
    const int lane = tid & 31, wid = tid >> 5;
    const int waveM = wid & 3, waveN = wid >> 2;
    const int mblk = blockIdx.y * 128, nblk = blockIdx.x * 128;
    const int m0 = mblk + waveM * 32;
    const int nloc = waveN * 64;                     // wave's column base inside LDS tile
    const int l16 = lane & 15, hi = lane >> 4;

    v8f acc[2][4] = {};

    stage_B(W, ldsB[0], nblk, 0, K, tid);

    for (int k = 0; k < K; k += 32) {
        const int buf = (k >> 5) & 1;

        // A fragments for this step: issue early (overlap the async-wait/barrier)
        v16bf a0 = load_frag(A, K, m0,      k, lane);
        v16bf a1 = load_frag(A, K, m0 + 16, k, lane);
        if (k + 32 < K)
            __builtin_prefetch(A + (size_t)(m0 + l16) * K + k + 32, 0, 1);

        wait_async_stage();          // this wave's chunks of ldsB[buf] landed
        __syncthreads();             // everyone's chunks landed; prev readers of buf^1 done

        if (k + 32 < K)
            stage_B(W, ldsB[buf ^ 1], nblk, k + 32, K, tid);  // overlaps WMMAs below

        #pragma unroll
        for (int nf = 0; nf < 4; ++nf) {
            v16bf b = load_frag(ldsB[buf], 32, nloc + nf * 16, 0, lane);  // ds_load_b128 x2
            acc[0][nf] = WMMA_BF16(a0, b, acc[0][nf]);
            acc[1][nf] = WMMA_BF16(a1, b, acc[1][nf]);
        }
    }

    // C/D layout: lane l16 = column, lanes 16-31 carry rows 8..15, VGPR r = row (hi*8+r)
    #pragma unroll
    for (int mf = 0; mf < 2; ++mf) {
        #pragma unroll
        for (int nf = 0; nf < 4; ++nf) {
            const int col = nblk + nloc + nf * 16 + l16;
            const float bv = bias[col];
            #pragma unroll
            for (int r = 0; r < 8; ++r) {
                const int row = m0 + mf * 16 + hi * 8 + r;
                const float v = acc[mf][nf][r] + bv;
                if (mode == 0)      ((bf16*)out)[(size_t)row * N + col] = (bf16)v;
                else if (mode == 1) ((bf16*)out)[(size_t)col * M + row] = (bf16)v; // Vt
                else                ((float*)out)[(size_t)row * N + col] = v;
            }
        }
    }
}

// ---------------------------------------------------------------------------
// Flash attention: one wave handles (batch b, head h, 16-query tile).
// Q,K: bf16 [4096,1024] row-major.  Vt: bf16 [1024,4096] (transposed).
__global__ void mha_flash_attn(const bf16* __restrict__ Qb, const bf16* __restrict__ Kb,
                               const bf16* __restrict__ Vt, bf16* __restrict__ Cb) {
    __shared__ alignas(16) bf16 ldsP[8][512]; // per-wave 16x32 probability tile

    const int lane = threadIdx.x & 31, wid = threadIdx.x >> 5;
    const int task = blockIdx.x * 8 + wid;          // 4096 tasks total
    const int st = task & 127;                      // query tile (2048/16)
    const int h  = (task >> 7) & 15;                // head
    const int b  = task >> 11;                      // batch
    const int l16 = lane & 15, hi = lane >> 4;

    const int rowq  = b * 2048 + st * 16;
    const int tbase = b * 2048;
    const int dcol  = h * 64;

    v16bf q0 = load_frag(Qb, 1024, rowq, dcol, lane);
    v16bf q1 = load_frag(Qb, 1024, rowq, dcol + 32, lane);

    float run_max[8], run_sum[8];
    #pragma unroll
    for (int r = 0; r < 8; ++r) { run_max[r] = -1e30f; run_sum[r] = 0.0f; }
    v8f acc[4] = {};

    const bf16* Kbat = Kb + (size_t)tbase * 1024;
    bf16* pl = ldsP[wid];

    for (int t0 = 0; t0 < 2048; t0 += 32) {
        // scores: two 16x16 tiles over t (B lane-column = K row t, contiguous)
        v16bf kb00 = load_frag(Kbat, 1024, t0,      dcol,      lane);
        v16bf kb01 = load_frag(Kbat, 1024, t0,      dcol + 32, lane);
        v16bf kb10 = load_frag(Kbat, 1024, t0 + 16, dcol,      lane);
        v16bf kb11 = load_frag(Kbat, 1024, t0 + 16, dcol + 32, lane);
        if (t0 + 32 < 2048) {
            __builtin_prefetch(Kbat + (size_t)(t0 + 32 + l16) * 1024 + dcol, 0, 1);
            __builtin_prefetch(Kbat + (size_t)(t0 + 48 + l16) * 1024 + dcol, 0, 1);
        }
        v8f s0 = {}, s1 = {};
        s0 = WMMA_BF16(q0, kb00, s0); s0 = WMMA_BF16(q1, kb01, s0);
        s1 = WMMA_BF16(q0, kb10, s1); s1 = WMMA_BF16(q1, kb11, s1);

        // online softmax; rows live across the 16-lane group (xor masks 1..8 stay in group)
        #pragma unroll
        for (int r = 0; r < 8; ++r) {
            const float a0 = s0[r] * 0.125f, a1 = s1[r] * 0.125f; // 1/sqrt(64)
            const float mnew = fmaxf(run_max[r], red_max16(fmaxf(a0, a1)));
            const float fac  = __expf(run_max[r] - mnew);
            const float p0 = __expf(a0 - mnew), p1 = __expf(a1 - mnew);
            run_sum[r] = run_sum[r] * fac + red_sum16(p0 + p1);
            run_max[r] = mnew;
            acc[0][r] *= fac; acc[1][r] *= fac; acc[2][r] *= fac; acc[3][r] *= fac;
            const int rowm = hi * 8 + r;                 // C-layout row for this lane
            pl[rowm * 32 + l16]      = (bf16)p0;
            pl[rowm * 32 + 16 + l16] = (bf16)p1;
        }
        asm volatile("s_wait_dscnt 0" ::: "memory");     // LDS RAW fence (wave-local)

        // P(16x32) @ V(32x64): A from LDS in A-layout, B lane-col = Vt row (contiguous)
        v16bf pa = load_frag(pl, 32, 0, 0, lane);
        #pragma unroll
        for (int f = 0; f < 4; ++f) {
            v16bf vb = load_frag(Vt, 4096, dcol + f * 16, tbase + t0, lane);
            acc[f] = WMMA_BF16(pa, vb, acc[f]);
        }
    }

    #pragma unroll
    for (int r = 0; r < 8; ++r) {
        const float inv = 1.0f / run_sum[r];
        const size_t row = (size_t)(rowq + hi * 8 + r);
        #pragma unroll
        for (int f = 0; f < 4; ++f)
            Cb[row * 1024 + dcol + f * 16 + l16] = (bf16)(acc[f][r] * inv);
    }
}

// ---------------------------------------------------------------------------
extern "C" void kernel_launch(void* const* d_in, const int* in_sizes, int n_in,
                              void* d_out, int out_size, void* d_ws, size_t ws_size,
                              hipStream_t stream) {
    const float* x  = (const float*)d_in[0];
    const float* wq = (const float*)d_in[1];
    const float* bq = (const float*)d_in[2];
    const float* wk = (const float*)d_in[3];
    const float* bk = (const float*)d_in[4];
    const float* wv = (const float*)d_in[5];
    const float* bv = (const float*)d_in[6];
    const float* wo = (const float*)d_in[7];
    const float* bo = (const float*)d_in[8];

    const int M = 4096, D = 1024;          // M = 2*2048 tokens
    const size_t MB = 1u << 20;
    char* ws = (char*)d_ws;
    bf16* xb  = (bf16*)(ws + 0 * MB);      // 8 MB  [4096,1024]
    bf16* wqb = (bf16*)(ws + 8 * MB);      // 2 MB
    bf16* wkb = (bf16*)(ws + 10 * MB);
    bf16* wvb = (bf16*)(ws + 12 * MB);
    bf16* wob = (bf16*)(ws + 14 * MB);
    bf16* Qb  = (bf16*)(ws + 16 * MB);     // 8 MB
    bf16* Kb  = (bf16*)(ws + 24 * MB);     // 8 MB
    bf16* Vtb = (bf16*)(ws + 32 * MB);     // 8 MB  [1024,4096] transposed
    bf16* Cb  = (bf16*)(ws + 40 * MB);     // 8 MB  context

    const int nX = M * D, nW = D * D;
    mha_cvt_bf16<<<nX / (256 * 8), 256, 0, stream>>>(x,  xb,  nX);
    mha_cvt_bf16<<<nW / (256 * 8), 256, 0, stream>>>(wq, wqb, nW);
    mha_cvt_bf16<<<nW / (256 * 8), 256, 0, stream>>>(wk, wkb, nW);
    mha_cvt_bf16<<<nW / (256 * 8), 256, 0, stream>>>(wv, wvb, nW);
    mha_cvt_bf16<<<nW / (256 * 8), 256, 0, stream>>>(wo, wob, nW);

    dim3 ggrid(D / 128, M / 128), gblk(256);
    mha_gemm_wmma<<<ggrid, gblk, 0, stream>>>(xb, wqb, bq, Qb,  M, D, D, 0);
    mha_gemm_wmma<<<ggrid, gblk, 0, stream>>>(xb, wkb, bk, Kb,  M, D, D, 0);
    mha_gemm_wmma<<<ggrid, gblk, 0, stream>>>(xb, wvb, bv, Vtb, M, D, D, 1); // transposed

    mha_flash_attn<<<512, 256, 0, stream>>>(Qb, Kb, Vtb, Cb);

    mha_gemm_wmma<<<ggrid, gblk, 0, stream>>>(Cb, wob, bo, d_out, M, D, D, 2); // f32 out
}
